// TestSparseMoeBlock_7645041787398
// MI455X (gfx1250) — matrix operand order repaired and checked by reading, not compiled, not run on hardware
//
#include <hip/hip_runtime.h>

#define H_DIM 2048
#define F_DIM 8192
#define E_NUM 8
#define T_NUM 512      // B*S tokens
#define TM 128
#define TN 128
#define TK 32
#define LDSS 40        // padded bf16 stride: 80B rows -> 16B-aligned frags

typedef __attribute__((ext_vector_type(16))) __bf16 v16bf;
typedef __attribute__((ext_vector_type(4)))  __bf16 v4bf;
typedef __attribute__((ext_vector_type(8)))  float  v8f;
typedef __attribute__((ext_vector_type(4)))  float  v4f;
typedef __attribute__((ext_vector_type(4)))  unsigned int u32x4;

union FragU { u32x4 q[2]; v16bf v; };

__device__ __forceinline__ v16bf frag_from(const __bf16* p0, const __bf16* p1) {
    FragU f;
    f.q[0] = *(const u32x4*)p0;   // 16B chunk: elements 0..7
    f.q[1] = *(const u32x4*)p1;   // 16B chunk: elements 8..15
    return f.v;
}

__device__ __forceinline__ v4bf pack4(v4f v) {
    v4bf p; p[0] = (__bf16)v[0]; p[1] = (__bf16)v[1];
            p[2] = (__bf16)v[2]; p[3] = (__bf16)v[3];
    return p;
}

// ---------------------------------------------------------------- router ----
__global__ __launch_bounds__(256) void moe_router(
    const float* __restrict__ x, const float* __restrict__ Wg,
    int* __restrict__ counts, int* __restrict__ list, float* __restrict__ wt)
{
    int t = blockIdx.x * 256 + threadIdx.x;
    if (t >= T_NUM) return;

    float acc[E_NUM];
#pragma unroll
    for (int e = 0; e < E_NUM; ++e) acc[e] = 0.f;

    const float* xr = x + (size_t)t * H_DIM;
    for (int h = 0; h < H_DIM; h += 4) {
        v4f xv = *(const v4f*)&xr[h];
#pragma unroll
        for (int j = 0; j < 4; ++j) {
            v4f g0 = *(const v4f*)&Wg[(h + j) * E_NUM];
            v4f g1 = *(const v4f*)&Wg[(h + j) * E_NUM + 4];
#pragma unroll
            for (int e = 0; e < 4; ++e) { acc[e] += xv[j] * g0[e]; acc[4 + e] += xv[j] * g1[e]; }
        }
    }

    // top-2; renormalized top-2 softmax == softmax over the two top logits
    float best = -1e30f; int bi = 0;
#pragma unroll
    for (int e = 0; e < E_NUM; ++e) if (acc[e] > best) { best = acc[e]; bi = e; }
    float sec = -1e30f; int si = 0;
#pragma unroll
    for (int e = 0; e < E_NUM; ++e) if (e != bi && acc[e] > sec) { sec = acc[e]; si = e; }

    float w1 = 1.f / (1.f + __expf(sec - best));
    float w2 = 1.f - w1;

    int s1 = atomicAdd(&counts[bi], 1);
    list[bi * T_NUM + s1] = t;  wt[bi * T_NUM + s1] = w1;
    int s2 = atomicAdd(&counts[si], 1);
    list[si * T_NUM + s2] = t;  wt[si * T_NUM + s2] = w2;
}

// ------------------------------------------------- pass A: h = relu(xW1)*xW3
__global__ __launch_bounds__(256) void moe_ffn_up(
    const float* __restrict__ x, const float* __restrict__ W1,
    const float* __restrict__ W3, const int* __restrict__ counts,
    const int* __restrict__ list, __bf16* __restrict__ Hact)
{
    const int e    = blockIdx.z;
    const int mblk = blockIdx.y;
    const int f0   = blockIdx.x * TN;
    const int cnt  = counts[e];
    if (mblk * TM >= cnt) return;

    __shared__ __bf16 sA [2][TM * LDSS];   // gathered X tile, row-major [m][k]
    __shared__ __bf16 sW1[2][TN * LDSS];   // W1 tile, transposed [n][k]
    __shared__ __bf16 sW3[2][TN * LDSS];

    const int tid  = threadIdx.x;
    const int lane = tid & 31, wv = tid >> 5;
    const int l16  = lane & 15, half = lane >> 4;

    const float* W1e = W1 + (size_t)e * H_DIM * F_DIM;
    const float* W3e = W3 + (size_t)e * H_DIM * F_DIM;
    const int*   le  = list + e * T_NUM;

    // --- per-thread staging descriptors (hoisted; __syncthreads blocks LICM) ---
    const float* xptr[4];  int offA[4];        // X gather: 4 float4 chunks
    size_t gofs[4];        int offW[4];        // W tiles:  4 float4 chunks each
#pragma unroll
    for (int i = 0; i < 4; ++i) {
        int idx = tid + i * 256;
        int k4 = idx & 7, r = idx >> 3;        // 8 chunks per A row
        int s = mblk * TM + r;
        int tok = (s < cnt) ? le[s] : 0;
        xptr[i] = x + (size_t)tok * H_DIM + k4 * 4;
        offA[i] = r * LDSS + k4 * 4;
        int n4 = idx & 31, k = idx >> 5;       // 32 chunks per W k-row
        gofs[i] = (size_t)k * F_DIM + f0 + n4 * 4;
        offW[i] = (n4 * 4) * LDSS + k;
    }

    v8f acc1[8], acc3[8];
    v8f zf = {0.f,0.f,0.f,0.f,0.f,0.f,0.f,0.f};
#pragma unroll
    for (int i = 0; i < 8; ++i) { acc1[i] = zf; acc3[i] = zf; }

    v4f ra[4], rw1[4], rw3[4];
    // prologue: load tile 0 and park it in LDS buffer 0
#pragma unroll
    for (int i = 0; i < 4; ++i) {
        ra [i] = *(const v4f*)(xptr[i]);
        rw1[i] = *(const v4f*)(W1e + gofs[i]);
        rw3[i] = *(const v4f*)(W3e + gofs[i]);
    }
#pragma unroll
    for (int i = 0; i < 4; ++i) {
        *(v4bf*)&sA[0][offA[i]] = pack4(ra[i]);
#pragma unroll
        for (int j = 0; j < 4; ++j) {
            sW1[0][offW[i] + j * LDSS] = (__bf16)rw1[i][j];
            sW3[0][offW[i] + j * LDSS] = (__bf16)rw3[i][j];
        }
    }

    const int NSTEP = H_DIM / TK;   // 64
    const int c = wv * 16 + l16;    // this wave's B column
    for (int it = 0; it < NSTEP; ++it) {
        const int cb = it & 1;
        __syncthreads();            // tile(it) visible; prior reads of other buf done
        // issue next tile's global loads -> overlap with WMMA below
        if (it + 1 < NSTEP) {
            const int k0n = (it + 1) * TK;
#pragma unroll
            for (int i = 0; i < 4; ++i) {
                ra [i] = *(const v4f*)(xptr[i] + k0n);
                rw1[i] = *(const v4f*)(W1e + gofs[i] + (size_t)k0n * F_DIM);
                rw3[i] = *(const v4f*)(W3e + gofs[i] + (size_t)k0n * F_DIM);
            }
        }
        // compute from LDS buffer cb
        v16bf b1 = frag_from(&sW1[cb][c * LDSS + half * 8], &sW1[cb][c * LDSS + 16 + half * 8]);
        v16bf b3 = frag_from(&sW3[cb][c * LDSS + half * 8], &sW3[cb][c * LDSS + 16 + half * 8]);
#pragma unroll
        for (int mt = 0; mt < 8; ++mt) {
            int m = mt * 16 + l16;
            v16bf a = frag_from(&sA[cb][m * LDSS + half * 8], &sA[cb][m * LDSS + 16 + half * 8]);
            acc1[mt] = __builtin_amdgcn_wmma_f32_16x16x32_bf16(
                           false, a, false, b1, (short)0, acc1[mt], false, false);
            acc3[mt] = __builtin_amdgcn_wmma_f32_16x16x32_bf16(
                           false, a, false, b3, (short)0, acc3[mt], false, false);
        }
        // park next tile in the other LDS buffer
        if (it + 1 < NSTEP) {
            const int nb = cb ^ 1;
#pragma unroll
            for (int i = 0; i < 4; ++i) {
                *(v4bf*)&sA[nb][offA[i]] = pack4(ra[i]);
#pragma unroll
                for (int j = 0; j < 4; ++j) {
                    sW1[nb][offW[i] + j * LDSS] = (__bf16)rw1[i][j];
                    sW3[nb][offW[i] + j * LDSS] = (__bf16)rw3[i][j];
                }
            }
        }
    }

    // fused gate epilogue -> bf16 activation buffer
    const int n = f0 + wv * 16 + l16;
#pragma unroll
    for (int mt = 0; mt < 8; ++mt) {
#pragma unroll
        for (int r = 0; r < 8; ++r) {
            int s = mblk * TM + mt * 16 + half * 8 + r;
            if (s < cnt) {
                float hv = fmaxf(acc1[mt][r], 0.f) * acc3[mt][r];
                Hact[(size_t)(e * T_NUM + s) * F_DIM + n] = (__bf16)hv;
            }
        }
    }
}

// --------------------------------------------- pass B: y = h W2, scatter-add
__global__ __launch_bounds__(256) void moe_ffn_down(
    const __bf16* __restrict__ Hact, const float* __restrict__ W2,
    const int* __restrict__ counts, const int* __restrict__ list,
    const float* __restrict__ wt, float* __restrict__ out)
{
    const int e    = blockIdx.z;
    const int mblk = blockIdx.y;
    const int h0   = blockIdx.x * TN;
    const int cnt  = counts[e];
    if (mblk * TM >= cnt) return;

    __shared__ __bf16 sW[2][TN * LDSS];   // W2 tile transposed [n][k]

    const int tid  = threadIdx.x;
    const int lane = tid & 31, wv = tid >> 5;
    const int l16  = lane & 15, half = lane >> 4;

    const float*  W2e = W2 + (size_t)e * F_DIM * H_DIM;
    const __bf16* Ae  = Hact + (size_t)e * T_NUM * F_DIM;

    // staging descriptors (hoisted)
    size_t gofs[4]; int offW[4];
#pragma unroll
    for (int i = 0; i < 4; ++i) {
        int idx = tid + i * 256;
        int n4 = idx & 31, k = idx >> 5;
        gofs[i] = (size_t)k * H_DIM + h0 + n4 * 4;
        offW[i] = (n4 * 4) * LDSS + k;
    }
    // per-lane A-fragment row pointers (one row per mt, shared by both halves)
    const __bf16* arow[8];
#pragma unroll
    for (int mt = 0; mt < 8; ++mt)
        arow[mt] = Ae + (size_t)(mblk * TM + mt * 16 + l16) * F_DIM + half * 8;

    v8f acc[8];
    v8f zf = {0.f,0.f,0.f,0.f,0.f,0.f,0.f,0.f};
#pragma unroll
    for (int i = 0; i < 8; ++i) acc[i] = zf;

    v4f rw[4];
    v16bf afr[8];
    // prologue: tile 0 weights -> LDS buf 0; A fragments for k0=0 -> regs
#pragma unroll
    for (int i = 0; i < 4; ++i) rw[i] = *(const v4f*)(W2e + gofs[i]);
#pragma unroll
    for (int mt = 0; mt < 8; ++mt)
        afr[mt] = frag_from(arow[mt], arow[mt] + 16);
#pragma unroll
    for (int i = 0; i < 4; ++i) {
#pragma unroll
        for (int j = 0; j < 4; ++j)
            sW[0][offW[i] + j * LDSS] = (__bf16)rw[i][j];
    }

    const int NSTEP = F_DIM / TK;   // 256
    const int c = wv * 16 + l16;
    for (int it = 0; it < NSTEP; ++it) {
        const int cb = it & 1;
        const bool more = (it + 1 < NSTEP);
        const int k0n = (it + 1) * TK;
        __syncthreads();
        if (more) {
#pragma unroll
            for (int i = 0; i < 4; ++i)
                rw[i] = *(const v4f*)(W2e + gofs[i] + (size_t)k0n * H_DIM);
        }
        v16bf b = frag_from(&sW[cb][c * LDSS + half * 8], &sW[cb][c * LDSS + 16 + half * 8]);
#pragma unroll
        for (int mt = 0; mt < 8; ++mt) {
            acc[mt] = __builtin_amdgcn_wmma_f32_16x16x32_bf16(
                          false, afr[mt], false, b, (short)0, acc[mt], false, false);
            if (more)  // refill this mt's A fragment for the next k-step
                afr[mt] = frag_from(arow[mt] + k0n, arow[mt] + k0n + 16);
        }
        if (more) {
            const int nb = cb ^ 1;
#pragma unroll
            for (int i = 0; i < 4; ++i) {
#pragma unroll
                for (int j = 0; j < 4; ++j)
                    sW[nb][offW[i] + j * LDSS] = (__bf16)rw[i][j];
            }
        }
    }

    const int n = h0 + wv * 16 + l16;
#pragma unroll
    for (int mt = 0; mt < 8; ++mt) {
#pragma unroll
        for (int r = 0; r < 8; ++r) {
            int s = mblk * TM + mt * 16 + half * 8 + r;
            if (s < cnt) {
                int   tok = list[e * T_NUM + s];
                float wgt = wt[e * T_NUM + s];
                atomicAdd(&out[(size_t)tok * H_DIM + n], wgt * acc[mt][r]);
            }
        }
    }
}

// -------------------------------------------------------------------- launch
extern "C" void kernel_launch(void* const* d_in, const int* in_sizes, int n_in,
                              void* d_out, int out_size, void* d_ws, size_t ws_size,
                              hipStream_t stream)
{
    (void)in_sizes; (void)n_in; (void)ws_size;
    const float* x  = (const float*)d_in[0];
    const float* Wg = (const float*)d_in[1];
    const float* W1 = (const float*)d_in[2];
    const float* W2 = (const float*)d_in[3];
    const float* W3 = (const float*)d_in[4];
    float* out = (float*)d_out;

    char* ws = (char*)d_ws;
    int*    counts = (int*)ws;                                  // 8 ints
    int*    list   = (int*)(ws + 256);                          // E*T ints
    float*  wt     = (float*)(ws + 256 + E_NUM * T_NUM * 4);    // E*T floats
    __bf16* Hact   = (__bf16*)(ws + 65536);                     // E*T*F bf16 = 64 MB

    hipMemsetAsync(counts, 0, E_NUM * sizeof(int), stream);
    hipMemsetAsync(out, 0, (size_t)out_size * sizeof(float), stream);

    moe_router<<<dim3((T_NUM + 255) / 256), dim3(256), 0, stream>>>(
        x, Wg, counts, list, wt);
    moe_ffn_up<<<dim3(F_DIM / TN, T_NUM / TM, E_NUM), dim3(256), 0, stream>>>(
        x, W1, W3, counts, list, Hact);
    moe_ffn_down<<<dim3(H_DIM / TN, T_NUM / TM, E_NUM), dim3(256), 0, stream>>>(
        Hact, W2, counts, list, wt, out);
}
